// LlamaAttention_27453430956840
// MI455X (gfx1250) — matrix operand order repaired and checked by reading, not compiled, not run on hardware
//
#include <hip/hip_runtime.h>

// LlamaAttention forward for gfx1250 (MI455X): f16 WMMA GEMMs + flash attention.
// d_out = [out (B*S*H) | k (B*NH*S*HD) | v (B*NH*S*HD)] fp32.

typedef __attribute__((ext_vector_type(16))) _Float16 v16h;
typedef __attribute__((ext_vector_type(8)))  _Float16 v8h;
typedef __attribute__((ext_vector_type(8)))  float    v8f;
typedef __attribute__((ext_vector_type(4)))  int      v4i;

union V16u { v16h v; v8h h[2]; };

static constexpr int Bc = 2, Sc = 2048, Hc = 2048, NHc = 16, HDc = 128;

#if defined(__gfx1250__) && __has_builtin(__builtin_amdgcn_global_load_async_to_lds_b128)
#define HAVE_ASYNC_LDS 1
#else
#define HAVE_ASYNC_LDS 0
#endif

#if HAVE_ASYNC_LDS
typedef __attribute__((address_space(1))) v4i gv4i;   // global int4
typedef __attribute__((address_space(3))) v4i lv4i;   // LDS int4
#endif

// 16-byte global -> LDS copy (per-lane addresses). Async DMA path on CDNA5.
__device__ __forceinline__ void copy16B_g2l(const _Float16* g, _Float16* l) {
#if HAVE_ASYNC_LDS
  __builtin_amdgcn_global_load_async_to_lds_b128((gv4i*)g, (lv4i*)l, 0, 0);
#else
  *(v8h*)l = *(const v8h*)g;
#endif
}
__device__ __forceinline__ void wait_async0() {
#if HAVE_ASYNC_LDS
#if __has_builtin(__builtin_amdgcn_s_wait_asynccnt)
  __builtin_amdgcn_s_wait_asynccnt(0);
#else
  asm volatile("s_wait_asynccnt 0x0" ::: "memory");
#endif
#endif
}

// ---- fragment loaders (NT GEMM: both operands stored row-major, K contiguous) ----
// A fragment 16x32 f16: lane L(0-15): row M=L, K 0-7 & 16-23 ; lanes 16-31: K 8-15 & 24-31
__device__ __forceinline__ v16h frag_a_g(const _Float16* m, int row0, int col0, int ld) {
  int lane = threadIdx.x & 31;
  int r = lane & 15, s = lane >> 4;
  const _Float16* p = m + (size_t)(row0 + r) * ld + col0 + s * 8;
  V16u u; u.h[0] = *(const v8h*)p; u.h[1] = *(const v8h*)(p + 16);
  return u.v;
}
// B fragment 32x16 f16 from matrix stored m[n*ld + k]: lane L(0-15): col N=L, K 0-7 & 16-23;
// lanes 16-31: K 8-15 & 24-31
__device__ __forceinline__ v16h frag_b_g(const _Float16* m, int nrow0, int k0, int ld) {
  int lane = threadIdx.x & 31;
  int r = lane & 15, s = lane >> 4;
  const _Float16* p = m + (size_t)(nrow0 + r) * ld + k0 + s * 8;
  V16u u; u.h[0] = *(const v8h*)p; u.h[1] = *(const v8h*)(p + 16);
  return u.v;
}

__device__ __forceinline__ float half_max(float v) {
  v = fmaxf(v, __shfl_xor(v, 1, 32));
  v = fmaxf(v, __shfl_xor(v, 2, 32));
  v = fmaxf(v, __shfl_xor(v, 4, 32));
  v = fmaxf(v, __shfl_xor(v, 8, 32));
  return v;
}
__device__ __forceinline__ float half_sum(float v) {
  v += __shfl_xor(v, 1, 32);
  v += __shfl_xor(v, 2, 32);
  v += __shfl_xor(v, 4, 32);
  v += __shfl_xor(v, 8, 32);
  return v;
}

// ---- fp32 -> f16 conversion ----
__global__ void cvt_f32_f16(const float* __restrict__ in, _Float16* __restrict__ out, size_t n) {
  size_t i = (size_t)blockIdx.x * blockDim.x + threadIdx.x;
  size_t st = (size_t)gridDim.x * blockDim.x;
  for (; i < n; i += st) out[i] = (_Float16)in[i];
}

// ---- NT GEMM: C[M,N] = A[M,K] * W[N,K]^T, f16 in, f32 out. ----
// Block tile 128x128, 8 waves (2x4 -> 64x32 per wave), double-buffered LDS panels
// filled by async global->LDS DMA overlapped with WMMA on the other buffer.
__global__ __launch_bounds__(256) void gemm_nt_f16(const _Float16* __restrict__ A,
                                                   const _Float16* __restrict__ W,
                                                   float* __restrict__ C,
                                                   int M, int N, int K) {
  __shared__ _Float16 sA[2][128 * 40];   // 32-col panel, padded stride 40 halves (80B)
  __shared__ _Float16 sB[2][128 * 40];
  int tid = threadIdx.x;
  int wave = tid >> 5, lane = tid & 31;
  int wr = wave >> 2, wc = wave & 3;
  int row0 = blockIdx.y * 128;
  int col0 = blockIdx.x * 128;
  int r = lane & 15, s = lane >> 4;
  int srow = tid >> 1;                    // staging: row 0..127
  int shalf = (tid & 1) * 16;             // 16-half segment within the 32-wide panel
  v8f acc[4][2] = {};

  const _Float16* gA = A + (size_t)(row0 + srow) * K + shalf;
  const _Float16* gW = W + (size_t)(col0 + srow) * K + shalf;
  int lofs = srow * 40 + shalf;

  // prime buffer 0
  copy16B_g2l(gA,     &sA[0][lofs]);
  copy16B_g2l(gA + 8, &sA[0][lofs + 8]);
  copy16B_g2l(gW,     &sB[0][lofs]);
  copy16B_g2l(gW + 8, &sB[0][lofs + 8]);

  for (int k0 = 0; k0 < K; k0 += 32) {
    int cur = (k0 >> 5) & 1;
    wait_async0();
    __syncthreads();                      // panels for this chunk visible block-wide
    if (k0 + 32 < K) {                    // kick off DMA for the next chunk
      int nxt = cur ^ 1;
      const _Float16* ga = gA + k0 + 32;
      const _Float16* gw = gW + k0 + 32;
      copy16B_g2l(ga,     &sA[nxt][lofs]);
      copy16B_g2l(ga + 8, &sA[nxt][lofs + 8]);
      copy16B_g2l(gw,     &sB[nxt][lofs]);
      copy16B_g2l(gw + 8, &sB[nxt][lofs + 8]);
    }
    v16h bf[2];
#pragma unroll
    for (int ni = 0; ni < 2; ni++) {
      const _Float16* p = &sB[cur][(wc * 32 + ni * 16 + r) * 40 + s * 8];
      V16u u; u.h[0] = *(const v8h*)p; u.h[1] = *(const v8h*)(p + 16);
      bf[ni] = u.v;
    }
#pragma unroll
    for (int mi = 0; mi < 4; mi++) {
      const _Float16* p = &sA[cur][(wr * 64 + mi * 16 + r) * 40 + s * 8];
      V16u u; u.h[0] = *(const v8h*)p; u.h[1] = *(const v8h*)(p + 16);
      v16h af = u.v;
      acc[mi][0] = __builtin_amdgcn_wmma_f32_16x16x32_f16(false, af, false, bf[0], (short)0, acc[mi][0], false, false);
      acc[mi][1] = __builtin_amdgcn_wmma_f32_16x16x32_f16(false, af, false, bf[1], (short)0, acc[mi][1], false, false);
    }
  }
  // C/D layout: lane half s: rows i+8*s, col = r
#pragma unroll
  for (int mi = 0; mi < 4; mi++)
#pragma unroll
    for (int ni = 0; ni < 2; ni++) {
      int rb = row0 + wr * 64 + mi * 16 + s * 8;
      int c = col0 + wc * 32 + ni * 16 + r;
#pragma unroll
      for (int i = 0; i < 8; i++) C[(size_t)(rb + i) * N + c] = acc[mi][ni][i];
    }
}

// ---- RoPE: q/k fp32 (b*s, h) -> q16/k16 (b,h,s,d) f16 (+ k fp32 to d_out). q pre-scaled. ----
__global__ void rope_kernel(const float* __restrict__ qf, const float* __restrict__ kf,
                            const int* __restrict__ pos,
                            _Float16* __restrict__ q16, _Float16* __restrict__ k16,
                            float* __restrict__ k_out) {
  int t = blockIdx.x * blockDim.x + threadIdx.x;  // exactly B*S*NH*64 threads
  int d = t & 63;
  int nh = (t >> 6) & 15;
  int srow = t >> 10;                              // b*S + s
  int b = srow >> 11, sp = srow & 2047;
  float p = (float)pos[srow];
  const float ldc = -9.210340371976184f / 64.0f;   // -ln(10000)/(HD/2)
  float ang = p * __expf(ldc * (float)d);
  float sn, cs; __sincosf(ang, &sn, &cs);
  size_t gi = (size_t)srow * Hc + nh * HDc + d;
  float q0v = qf[gi], q1v = qf[gi + 64];
  float k0v = kf[gi], k1v = kf[gi + 64];
  float qr0 = q0v * cs - q1v * sn;
  float qr1 = q1v * cs + q0v * sn;
  float kr0 = k0v * cs - k1v * sn;
  float kr1 = k1v * cs + k0v * sn;
  const float scq = 0.08838834764831845f;          // 1/sqrt(HD)
  size_t oi = (((size_t)(b * NHc + nh)) * Sc + sp) * HDc + d;
  q16[oi] = (_Float16)(qr0 * scq);
  q16[oi + 64] = (_Float16)(qr1 * scq);
  k16[oi] = (_Float16)kr0;
  k16[oi + 64] = (_Float16)kr1;
  k_out[oi] = kr0;
  k_out[oi + 64] = kr1;
}

// ---- V pack: fp32 (b*s,h) -> d_out (b,h,s,d) fp32 + vT (b,h,d,s) f16 ----
__global__ void vpack_kernel(const float* __restrict__ vf, float* __restrict__ v_out,
                             _Float16* __restrict__ vT) {
  int t = blockIdx.x * blockDim.x + threadIdx.x;   // exactly B*S*H threads
  int d = t & 127;
  int nh = (t >> 7) & 15;
  int srow = t >> 11;
  int b = srow >> 11, sp = srow & 2047;
  float val = vf[(size_t)srow * Hc + nh * HDc + d];
  size_t bh = (size_t)(b * NHc + nh);
  v_out[(bh * Sc + sp) * HDc + d] = val;
  vT[(bh * HDc + d) * Sc + sp] = (_Float16)val;
}

// ---- Flash attention (causal). One 16-row q tile per wave; 8 waves per block. ----
__global__ __launch_bounds__(256) void flash_attn(const _Float16* __restrict__ qf,
                                                  const _Float16* __restrict__ kf,
                                                  const _Float16* __restrict__ vT,
                                                  _Float16* __restrict__ attn) {
  __shared__ _Float16 sP[8][16 * 40];              // per-wave 16x32 P bounce tile
  int tid = threadIdx.x;
  int wave = tid >> 5, lane = tid & 31;
  int r = lane & 15, sH = lane >> 4;
  int b = blockIdx.z, h = blockIdx.y;
  int q0 = blockIdx.x * 128 + wave * 16;
  size_t bh = (size_t)(b * NHc + h);
  const _Float16* qbase = qf + bh * (size_t)Sc * HDc;
  const _Float16* kbase = kf + bh * (size_t)Sc * HDc;
  const _Float16* vbase = vT + bh * (size_t)HDc * Sc;

  v16h qfr[4];
#pragma unroll
  for (int dj = 0; dj < 4; dj++) qfr[dj] = frag_a_g(qbase, q0, dj * 32, HDc);

  v8f acc[8] = {};
  float mrow[8], lrow[8];
#pragma unroll
  for (int i = 0; i < 8; i++) { mrow[i] = -1e30f; lrow[i] = 0.0f; }

  for (int kc = 0; kc < q0 + 16; kc += 32) {       // causal: skip tiles past diagonal
    v8f s0 = {}, s1 = {};
#pragma unroll
    for (int dj = 0; dj < 4; dj++) {
      v16h kb0 = frag_b_g(kbase, kc,      dj * 32, HDc);
      v16h kb1 = frag_b_g(kbase, kc + 16, dj * 32, HDc);
      s0 = __builtin_amdgcn_wmma_f32_16x16x32_f16(false, qfr[dj], false, kb0, (short)0, s0, false, false);
      s1 = __builtin_amdgcn_wmma_f32_16x16x32_f16(false, qfr[dj], false, kb1, (short)0, s1, false, false);
    }
    if (kc + 31 > q0) {                             // wave-uniform; per-element selects
#pragma unroll
      for (int i = 0; i < 8; i++) {
        int qrow = q0 + sH * 8 + i;
        s0[i] = (qrow >= kc + r)      ? s0[i] : -1e30f;
        s1[i] = (qrow >= kc + 16 + r) ? s1[i] : -1e30f;
      }
    }
    float p0[8], p1[8], alpha[8];
#pragma unroll
    for (int i = 0; i < 8; i++) {
      float mx = half_max(fmaxf(s0[i], s1[i]));
      float nm = fmaxf(mrow[i], mx);
      alpha[i] = __expf(mrow[i] - nm);
      p0[i] = __expf(s0[i] - nm);
      p1[i] = __expf(s1[i] - nm);
      float rs = half_sum(p0[i] + p1[i]);
      lrow[i] = lrow[i] * alpha[i] + rs;
      mrow[i] = nm;
    }
#pragma unroll
    for (int f = 0; f < 8; f++)
#pragma unroll
      for (int i = 0; i < 8; i++) acc[f][i] = acc[f][i] * alpha[i];
    // D-layout -> A-layout transpose of P via per-wave LDS (no block barrier needed)
    _Float16* myP = &sP[wave][0];
#pragma unroll
    for (int i = 0; i < 8; i++) {
      myP[(sH * 8 + i) * 40 + r]      = (_Float16)p0[i];
      myP[(sH * 8 + i) * 40 + 16 + r] = (_Float16)p1[i];
    }
    V16u pu;
    pu.h[0] = *(const v8h*)&sP[wave][r * 40 + sH * 8];
    pu.h[1] = *(const v8h*)&sP[wave][r * 40 + sH * 8 + 16];
    v16h pfr = pu.v;
#pragma unroll
    for (int f = 0; f < 8; f++) {
      v16h vfr = frag_b_g(vbase, f * 16, kc, Sc);
      acc[f] = __builtin_amdgcn_wmma_f32_16x16x32_f16(false, pfr, false, vfr, (short)0, acc[f], false, false);
    }
  }
  float rl[8];
#pragma unroll
  for (int i = 0; i < 8; i++) rl[i] = 1.0f / lrow[i];
  // attn output (b, s, h*d) f16 for the O-projection GEMM
  _Float16* ob = attn + ((size_t)(b * Sc + q0 + sH * 8)) * Hc + h * HDc;
#pragma unroll
  for (int f = 0; f < 8; f++)
#pragma unroll
    for (int i = 0; i < 8; i++)
      ob[(size_t)i * Hc + f * 16 + r] = (_Float16)(acc[f][i] * rl[i]);
}

extern "C" void kernel_launch(void* const* d_in, const int* in_sizes, int n_in,
                              void* d_out, int out_size, void* d_ws, size_t ws_size,
                              hipStream_t stream) {
  const float* hs = (const float*)d_in[0];
  const int*   pos = (const int*)d_in[2];
  const float* Wq = (const float*)d_in[3];
  const float* Wk = (const float*)d_in[4];
  const float* Wv = (const float*)d_in[5];
  const float* Wo = (const float*)d_in[6];
  float* out_o = (float*)d_out;
  float* out_k = out_o + (size_t)Bc * Sc * Hc;
  float* out_v = out_k + (size_t)Bc * NHc * Sc * HDc;

  const size_t NE = (size_t)Bc * Sc * Hc;   // 8388608 activation elements
  const size_t WE = (size_t)Hc * Hc;        // 4194304 weight elements
  char* w = (char*)d_ws;
  _Float16* hs16 = (_Float16*)w; w += NE * 2;
  _Float16* wq16 = (_Float16*)w; w += WE * 2;
  _Float16* wk16 = (_Float16*)w; w += WE * 2;
  _Float16* wv16 = (_Float16*)w; w += WE * 2;
  _Float16* wo16 = (_Float16*)w; w += WE * 2;
  float* qf = (float*)w; w += NE * 4;
  float* kf = (float*)w; w += NE * 4;
  float* vf = (float*)w; w += NE * 4;
  _Float16* q16 = (_Float16*)w; w += NE * 2;
  _Float16* k16 = (_Float16*)w; w += NE * 2;
  _Float16* vT16 = (_Float16*)w; w += NE * 2;
  _Float16* attn16 = (_Float16*)qf;          // reuse dead q fp32 buffer

  cvt_f32_f16<<<4096, 256, 0, stream>>>(hs, hs16, NE);
  cvt_f32_f16<<<2048, 256, 0, stream>>>(Wq, wq16, WE);
  cvt_f32_f16<<<2048, 256, 0, stream>>>(Wk, wk16, WE);
  cvt_f32_f16<<<2048, 256, 0, stream>>>(Wv, wv16, WE);
  cvt_f32_f16<<<2048, 256, 0, stream>>>(Wo, wo16, WE);

  dim3 ggrid(Hc / 128, (Bc * Sc) / 128);     // (16, 32)
  gemm_nt_f16<<<ggrid, 256, 0, stream>>>(hs16, wq16, qf, Bc * Sc, Hc, Hc);
  gemm_nt_f16<<<ggrid, 256, 0, stream>>>(hs16, wk16, kf, Bc * Sc, Hc, Hc);
  gemm_nt_f16<<<ggrid, 256, 0, stream>>>(hs16, wv16, vf, Bc * Sc, Hc, Hc);

  rope_kernel<<<(Bc * Sc * NHc * 64) / 256, 256, 0, stream>>>(qf, kf, pos, q16, k16, out_k);
  vpack_kernel<<<(Bc * Sc * Hc) / 256, 256, 0, stream>>>(vf, out_v, vT16);

  dim3 fgrid(Sc / 128, NHc, Bc);             // (16, 16, 2)
  flash_attn<<<fgrid, 256, 0, stream>>>(q16, k16, vT16, attn16);

  gemm_nt_f16<<<ggrid, 256, 0, stream>>>(attn16, wo16, out_o, Bc * Sc, Hc, Hc);
}
